// BiLSTM_CRF_30399778521783
// MI455X (gfx1250) — compile-verified, matrix operand order checked by reading, not run on hardware
//
#include <hip/hip_runtime.h>
#include <hip/hip_bf16.h>
#include <stdint.h>

// ---------------------------------------------------------------------------
// BiLSTM-CRF inference for MI455X (gfx1250, wave32).
//   1) gather+cvt embedding rows to bf16
//   2) cvt weights to bf16
//   3) xp = x @ W_ih^T + b  (both dirs)  -- WMMA bf16 GEMM, f32 accum,
//      double-buffered LDS tiles filled with global_load_async_to_lds_b128
//   4) persistent LSTM scan, 2 workgroups (fwd/bwd concurrently),
//      h/c/gates in LDS, W_hh streamed bf16 from L2
//   5) feats = [h_f|h_b] @ fc_W^T + fc_b -- WMMA bf16 GEMM
//   6) Viterbi: one wave32, tag==lane, trans in LDS, lane-0 backtrack
// ---------------------------------------------------------------------------

typedef __bf16 v16bf __attribute__((ext_vector_type(16)));
typedef __bf16 v8bf  __attribute__((ext_vector_type(8)));
typedef __bf16 v2bf  __attribute__((ext_vector_type(2)));
typedef float  v8f   __attribute__((ext_vector_type(8)));

#define S_LEN 8192
#define E_DIM 512
#define H_DIM 512
#define G_DIM 2048   /* 4*H */
#define HID   1024
#define NTAG  32
#define TAG_START 30
#define TAG_STOP  31
#define NEGINF (-10000.0f)

// ------------------------- workspace layout (bytes) ------------------------
static constexpr size_t OFF_XBF  = 0;                                      // S*E bf16
static constexpr size_t OFF_WIHF = OFF_XBF  + (size_t)S_LEN * E_DIM * 2;   // G*E bf16
static constexpr size_t OFF_WIHB = OFF_WIHF + (size_t)G_DIM * E_DIM * 2;
static constexpr size_t OFF_WHHF = OFF_WIHB + (size_t)G_DIM * E_DIM * 2;   // G*H bf16
static constexpr size_t OFF_WHHB = OFF_WHHF + (size_t)G_DIM * H_DIM * 2;
static constexpr size_t OFF_FCW  = OFF_WHHB + (size_t)G_DIM * H_DIM * 2;   // T*HID bf16
static constexpr size_t OFF_XPF  = OFF_FCW  + (size_t)NTAG  * HID  * 2;    // S*G f32
static constexpr size_t OFF_XPB  = OFF_XPF  + (size_t)S_LEN * G_DIM * 4;
static constexpr size_t OFF_HF   = OFF_XPB  + (size_t)S_LEN * G_DIM * 4;   // S*H bf16
static constexpr size_t OFF_HB   = OFF_HF   + (size_t)S_LEN * H_DIM * 2;
static constexpr size_t OFF_FEAT = OFF_HB   + (size_t)S_LEN * H_DIM * 2;   // S*T f32
static constexpr size_t OFF_BP   = OFF_FEAT + (size_t)S_LEN * NTAG * 4;    // S*T u8

__device__ __forceinline__ float sigf(float x) { return 1.0f / (1.0f + __expf(-x)); }

// -------- CDNA5 async global->LDS copy (ASYNCcnt-tracked, bypasses VGPRs) --
__device__ __forceinline__ void async_load_b128(const void* gsrc, void* ldst)
{
    // low 32 bits of a generic pointer to __shared__ == LDS byte offset
    unsigned loff = (unsigned)(unsigned long long)ldst;
    asm volatile("global_load_async_to_lds_b128 %0, %1, off"
                 :: "v"(loff), "v"(gsrc)
                 : "memory");
}
__device__ __forceinline__ void wait_async0()
{
    asm volatile("s_wait_asynccnt 0x0" ::: "memory");
}

// -------------------- embedding gather + f32->bf16 -------------------------
__global__ void gather_embed(const int* __restrict__ idx,
                             const float* __restrict__ embed,
                             __bf16* __restrict__ xbf)
{
    int i = blockIdx.x * blockDim.x + threadIdx.x;      // over S*E
    int s = i >> 9;
    int e = i & (E_DIM - 1);
    xbf[i] = (__bf16)embed[(size_t)idx[s] * E_DIM + e];
}

__global__ void cvt_bf16(const float* __restrict__ in, __bf16* __restrict__ out, int n)
{
    int i = blockIdx.x * blockDim.x + threadIdx.x;
    if (i < n) out[i] = (__bf16)in[i];
}

// -------------------- xp GEMM: C[M,N] = A[M,K] * W[N,K]^T + bias[N] --------
// BM=128, BN=128, BK=32; 256 threads = 8 waves (4x2), wave tile 32x64
// (2x4 WMMA).  Double-buffered LDS, tiles filled by async-to-LDS b128.
#define BM 128
#define BN 128
#define BK 32
__global__ __launch_bounds__(256) void gemm_xp(
    const __bf16* __restrict__ A,     // [M,K] bf16 (x)
    const __bf16* __restrict__ W,     // [N,K] bf16 (W_ih rows = gates)
    const float*  __restrict__ bias,  // [N]
    float* __restrict__ C,            // [M,N] f32
    int M, int N, int K)
{
    __shared__ __align__(16) __bf16 As[2][BM][BK];   // 16 KB
    __shared__ __align__(16) __bf16 Bs[2][BN][BK];   // 16 KB

    const int tid  = threadIdx.x;
    const int wave = tid >> 5;
    const int lane = tid & 31;
    const int wm   = wave >> 1;          // 0..3 : 32-row strip
    const int wn   = wave & 1;           // 0..1 : 64-col strip
    const int m0   = blockIdx.y * BM;
    const int n0   = blockIdx.x * BN;
    const int half = lane >> 4;
    const int l16  = lane & 15;

    v8f acc[2][4] = {};

    // 128x32 bf16 tile = 512 b128 chunks; 256 threads -> 2 chunks each (A and B)
    auto stage_async = [&](int k0, int buf) {
        #pragma unroll
        for (int c = 0; c < 2; ++c) {
            int chunk = tid + c * 256;          // 0..511
            int r  = chunk >> 2;                // 0..127
            int cb = (chunk & 3) * 8;           // 0,8,16,24
            async_load_b128(A + (size_t)(m0 + r) * K + k0 + cb, &As[buf][r][cb]);
            async_load_b128(W + (size_t)(n0 + r) * K + k0 + cb, &Bs[buf][r][cb]);
        }
    };

    stage_async(0, 0);
    wait_async0();
    __syncthreads();

    for (int k0 = 0; k0 < K; k0 += BK) {
        const int buf = (k0 >> 5) & 1;
        if (k0 + BK < K) stage_async(k0 + BK, buf ^ 1);   // overlap next fill

        // A fragments for this wave's two 16-row subtiles
        v16bf afr[2];
        #pragma unroll
        for (int tm = 0; tm < 2; ++tm) {
            int m = wm * 32 + tm * 16 + l16;
            #pragma unroll
            for (int v = 0; v < 8; ++v) {
                int kk = (v < 4) ? (half * 8 + 2 * v) : (16 + half * 8 + 2 * (v - 4));
                v2bf p = *(const v2bf*)&As[buf][m][kk];
                afr[tm][2 * v]     = p.x;
                afr[tm][2 * v + 1] = p.y;
            }
        }
        #pragma unroll
        for (int tn = 0; tn < 4; ++tn) {
            v16bf b;
            int n = wn * 64 + tn * 16 + l16;
            #pragma unroll
            for (int v = 0; v < 8; ++v) {
                int kk = (v < 4) ? (half * 8 + 2 * v) : (16 + half * 8 + 2 * (v - 4));
                v2bf p = *(const v2bf*)&Bs[buf][n][kk];
                b[2 * v]     = p.x;
                b[2 * v + 1] = p.y;
            }
            acc[0][tn] = __builtin_amdgcn_wmma_f32_16x16x32_bf16(
                false, afr[0], false, b, (short)0, acc[0][tn], false, false);
            acc[1][tn] = __builtin_amdgcn_wmma_f32_16x16x32_bf16(
                false, afr[1], false, b, (short)0, acc[1][tn], false, false);
        }

        wait_async0();      // own wave's async fills done
        __syncthreads();    // everyone's fills visible / reads of old buf done
    }

    // epilogue: D layout — lane l16 = column, VGPR v -> row v + 8*half
    #pragma unroll
    for (int tm = 0; tm < 2; ++tm)
        #pragma unroll
        for (int tn = 0; tn < 4; ++tn) {
            int n = n0 + wn * 64 + tn * 16 + l16;
            float bv = bias[n];
            #pragma unroll
            for (int v = 0; v < 8; ++v) {
                int m = m0 + wm * 32 + tm * 16 + v + 8 * half;
                C[(size_t)m * N + n] = acc[tm][tn][v] + bv;
            }
        }
}

// -------------------- persistent bidirectional LSTM scan -------------------
// grid=2 (dir), block=1024. thread t computes gates j=t and j=t+1024.
// Gate layout (PyTorch): [i | f | g | o], each H=512 wide.
__global__ __launch_bounds__(1024) void lstm_scan(
    const float* __restrict__ xp_f, const float* __restrict__ xp_b,
    const __bf16* __restrict__ Whh_f, const __bf16* __restrict__ Whh_b,
    __bf16* __restrict__ hf_out, __bf16* __restrict__ hb_out)
{
    const int dir = blockIdx.x;
    const float*  xp   = dir ? xp_b  : xp_f;
    const __bf16* W    = dir ? Whh_b : Whh_f;
    __bf16*       hout = dir ? hb_out : hf_out;

    __shared__ float h_sh[H_DIM];
    __shared__ float c_sh[H_DIM];
    __shared__ float g_sh[G_DIM];

    const int t  = threadIdx.x;
    const int j0 = t;
    const int j1 = t + 1024;
    const __bf16* Wr0 = W + (size_t)j0 * H_DIM;
    const __bf16* Wr1 = W + (size_t)j1 * H_DIM;

    if (t < H_DIM) { h_sh[t] = 0.0f; c_sh[t] = 0.0f; }
    __syncthreads();

    for (int step = 0; step < S_LEN; ++step) {
        const int s = dir ? (S_LEN - 1 - step) : step;
        const float* xpr = xp + (size_t)s * G_DIM;
        if (step + 1 < S_LEN) {
            const int sn = dir ? (S_LEN - 2 - step) : (step + 1);
            __builtin_prefetch(xp + (size_t)sn * G_DIM + j0, 0, 1);  // global_prefetch_b8
        }
        float a0 = xpr[j0];
        float a1 = xpr[j1];
        for (int k = 0; k < H_DIM; k += 8) {
            v8bf w0 = *(const v8bf*)(Wr0 + k);
            v8bf w1 = *(const v8bf*)(Wr1 + k);
            #pragma unroll
            for (int u = 0; u < 8; ++u) {
                float hv = h_sh[k + u];
                a0 += hv * (float)w0[u];
                a1 += hv * (float)w1[u];
            }
        }
        g_sh[j0] = a0;
        g_sh[j1] = a1;
        __syncthreads();
        if (t < H_DIM) {
            float ig = sigf(g_sh[t]);
            float fg = sigf(g_sh[H_DIM + t]);
            float gg = tanhf(g_sh[2 * H_DIM + t]);
            float og = sigf(g_sh[3 * H_DIM + t]);
            float c  = fg * c_sh[t] + ig * gg;
            float h  = og * tanhf(c);
            c_sh[t] = c;
            h_sh[t] = h;
            hout[(size_t)s * H_DIM + t] = (__bf16)h;
        }
        __syncthreads();
    }
}

// -------------------- feats GEMM: [S,1024] x [1024,32]^T + fc_b ------------
// block = 128 (4 waves); each wave owns a 16-row strip, two 16-col WMMA tiles.
__global__ __launch_bounds__(128) void gemm_feats(
    const __bf16* __restrict__ hf, const __bf16* __restrict__ hb,   // [S,512]
    const __bf16* __restrict__ fcW,                                  // [32,1024]
    const float*  __restrict__ fcb,                                  // [32]
    float* __restrict__ feats)                                       // [S,32]
{
    const int wave = threadIdx.x >> 5;
    const int lane = threadIdx.x & 31;
    const int half = lane >> 4;
    const int l16  = lane & 15;
    const int m0   = (blockIdx.x * 4 + wave) * 16;

    v8f acc0 = {};
    v8f acc1 = {};

    for (int k0 = 0; k0 < HID; k0 += 32) {
        const __bf16* Asrc = (k0 < H_DIM) ? hf : hb;
        const int kbase = (k0 < H_DIM) ? k0 : (k0 - H_DIM);
        v16bf a, b0, b1;
        const int m = m0 + l16;
        #pragma unroll
        for (int v = 0; v < 8; ++v) {
            int kk = (v < 4) ? (half * 8 + 2 * v) : (16 + half * 8 + 2 * (v - 4));
            v2bf pa = *(const v2bf*)(Asrc + (size_t)m * H_DIM + kbase + kk);
            a[2 * v] = pa.x;  a[2 * v + 1] = pa.y;
            v2bf p0 = *(const v2bf*)(fcW + (size_t)l16 * HID + k0 + kk);
            b0[2 * v] = p0.x; b0[2 * v + 1] = p0.y;
            v2bf p1 = *(const v2bf*)(fcW + (size_t)(16 + l16) * HID + k0 + kk);
            b1[2 * v] = p1.x; b1[2 * v + 1] = p1.y;
        }
        acc0 = __builtin_amdgcn_wmma_f32_16x16x32_bf16(false, a, false, b0, (short)0, acc0, false, false);
        acc1 = __builtin_amdgcn_wmma_f32_16x16x32_bf16(false, a, false, b1, (short)0, acc1, false, false);
    }
    #pragma unroll
    for (int v = 0; v < 8; ++v) {
        int m = m0 + v + 8 * half;
        feats[(size_t)m * NTAG + l16]      = acc0[v] + fcb[l16];
        feats[(size_t)m * NTAG + 16 + l16] = acc1[v] + fcb[16 + l16];
    }
}

// -------------------- Viterbi: one wave32, tag == lane ---------------------
__global__ __launch_bounds__(32) void viterbi(
    const float* __restrict__ feats,        // [S,32]
    const float* __restrict__ trans,        // [32,32] row=prev, col=next
    unsigned char* __restrict__ bp,         // [S,32]
    float* __restrict__ out)                // [1 + S]
{
    __shared__ float tr[NTAG * NTAG];
    __shared__ float prev[NTAG];
    __shared__ float cur[NTAG];
    const int lane = threadIdx.x;

    for (int i = lane; i < NTAG * NTAG; i += NTAG) tr[i] = trans[i];
    prev[lane] = (lane == TAG_START) ? 0.0f : NEGINF;
    __syncthreads();

    for (int s = 0; s < S_LEN; ++s) {
        float best = -3.4e38f;
        int bpi = 0;
        #pragma unroll 4
        for (int p = 0; p < NTAG; ++p) {
            float sc = prev[p] + tr[p * NTAG + lane];
            if (sc > best) { best = sc; bpi = p; }     // '>' keeps first argmax
        }
        bp[(size_t)s * NTAG + lane] = (unsigned char)bpi;
        cur[lane] = best + feats[(size_t)s * NTAG + lane];
        __syncthreads();
        prev[lane] = cur[lane];
        __syncthreads();
    }

    cur[lane] = prev[lane] + tr[lane * NTAG + TAG_STOP];
    __syncthreads();
    if (lane == 0) {
        float bs = cur[0];
        int bt = 0;
        for (int p = 1; p < NTAG; ++p)
            if (cur[p] > bs) { bs = cur[p]; bt = p; }
        out[0] = bs;
        int tag = bt;
        out[1 + (S_LEN - 1)] = (float)tag;
        for (int s2 = S_LEN - 1; s2 >= 1; --s2) {
            tag = bp[(size_t)s2 * NTAG + tag];
            out[s2] = (float)tag;     // path[s2-1]
        }
    }
}

// ---------------------------------------------------------------------------
extern "C" void kernel_launch(void* const* d_in, const int* in_sizes, int n_in,
                              void* d_out, int out_size, void* d_ws, size_t ws_size,
                              hipStream_t stream)
{
    (void)in_sizes; (void)n_in; (void)out_size; (void)ws_size;

    const int*   inputs = (const int*)  d_in[0];
    const float* embed  = (const float*)d_in[1];
    const float* W_ih_f = (const float*)d_in[2];
    const float* W_hh_f = (const float*)d_in[3];
    const float* b_f    = (const float*)d_in[4];
    const float* W_ih_b = (const float*)d_in[5];
    const float* W_hh_b = (const float*)d_in[6];
    const float* b_b    = (const float*)d_in[7];
    const float* fc_W   = (const float*)d_in[8];
    const float* fc_b   = (const float*)d_in[9];
    const float* trans  = (const float*)d_in[10];
    float* out = (float*)d_out;
    char*  ws  = (char*)d_ws;

    __bf16* xbf   = (__bf16*)(ws + OFF_XBF);
    __bf16* wihf  = (__bf16*)(ws + OFF_WIHF);
    __bf16* wihb  = (__bf16*)(ws + OFF_WIHB);
    __bf16* whhf  = (__bf16*)(ws + OFF_WHHF);
    __bf16* whhb  = (__bf16*)(ws + OFF_WHHB);
    __bf16* fcwb  = (__bf16*)(ws + OFF_FCW);
    float*  xpf   = (float*) (ws + OFF_XPF);
    float*  xpb   = (float*) (ws + OFF_XPB);
    __bf16* hfo   = (__bf16*)(ws + OFF_HF);
    __bf16* hbo   = (__bf16*)(ws + OFF_HB);
    float*  feats = (float*) (ws + OFF_FEAT);
    unsigned char* bpb = (unsigned char*)(ws + OFF_BP);

    // 1) gather + convert
    gather_embed<<<(S_LEN * E_DIM) / 256, 256, 0, stream>>>(inputs, embed, xbf);
    // 2) weight conversions
    const int nw = G_DIM * E_DIM;
    cvt_bf16<<<(nw + 255) / 256, 256, 0, stream>>>(W_ih_f, wihf, nw);
    cvt_bf16<<<(nw + 255) / 256, 256, 0, stream>>>(W_ih_b, wihb, nw);
    cvt_bf16<<<(nw + 255) / 256, 256, 0, stream>>>(W_hh_f, whhf, nw);
    cvt_bf16<<<(nw + 255) / 256, 256, 0, stream>>>(W_hh_b, whhb, nw);
    cvt_bf16<<<(NTAG * HID + 255) / 256, 256, 0, stream>>>(fc_W, fcwb, NTAG * HID);
    // 3) input projections (WMMA, async double-buffered tiles)
    dim3 gx(G_DIM / BN, S_LEN / BM);
    gemm_xp<<<gx, 256, 0, stream>>>(xbf, wihf, b_f, xpf, S_LEN, G_DIM, E_DIM);
    gemm_xp<<<gx, 256, 0, stream>>>(xbf, wihb, b_b, xpb, S_LEN, G_DIM, E_DIM);
    // 4) both LSTM directions concurrently
    lstm_scan<<<2, 1024, 0, stream>>>(xpf, xpb, whhf, whhb, hfo, hbo);
    // 5) emission features (WMMA)
    gemm_feats<<<S_LEN / 64, 128, 0, stream>>>(hfo, hbo, fcwb, fc_b, feats);
    // 6) Viterbi + backtrack
    viterbi<<<1, 32, 0, stream>>>(feats, trans, bpb, out);
}
